// AdaptiveRNNCell_78417512891225
// MI455X (gfx1250) — matrix operand order repaired and checked by reading, not compiled, not run on hardware
//
#include <hip/hip_runtime.h>
#include <hip/hip_bf16.h>

#define B_SZ   32768
#define H_SZ   512
#define L_SZ   10
#define THRESH 0.9f
#define ACT_EPS 1e-15f

// Toggle for the gfx1250 async load-to-LDS staging path (inline asm).
#define USE_ASYNC_LDS 1

typedef __attribute__((ext_vector_type(16))) __bf16 v16bf;
typedef __attribute__((ext_vector_type(8)))  __bf16 v8bf;
typedef __attribute__((ext_vector_type(4)))  __bf16 v4bf;
typedef __attribute__((ext_vector_type(8)))  float  v8f;

#if USE_ASYNC_LDS
// Copy 16 bytes global -> LDS via the gfx1250 async DMA path (ASYNCcnt-tracked).
// VDST = LDS byte offset (low 32 bits of the generic LDS pointer), VADDR = 64-bit
// global address.
__device__ __forceinline__ void async_cp16(const void* g, void* l) {
  unsigned loff = (unsigned)(size_t)l;   // wave-relative LDS offset
  asm volatile("global_load_async_to_lds_b128 %0, %1, off"
               :: "v"(loff), "v"(g)
               : "memory");
}
__device__ __forceinline__ void wait_async0() {
  asm volatile("s_wait_asynccnt 0" ::: "memory");
}
#endif

// Stage 16 rows (16*H bf16 = 16KB) of the persistent bf16 hidden state to LDS.
__device__ __forceinline__ void stage_h_bf(const __bf16* __restrict__ gsrc,
                                           __bf16* lds, int tid) {
#if USE_ASYNC_LDS
  for (int c = tid; c < 16 * H_SZ / 8; c += 128)
    async_cp16(gsrc + c * 8, lds + c * 8);
  wait_async0();
#else
  for (int c = tid; c < 16 * H_SZ / 8; c += 128)
    *(v8bf*)(lds + c * 8) = *(const v8bf*)(gsrc + c * 8);
#endif
}

// ---- WMMA helpers -----------------------------------------------------------

__device__ __forceinline__ v8f wmma_bf16(v16bf a, v16bf b, v8f c) {
  return __builtin_amdgcn_wmma_f32_16x16x32_bf16(
      false, a, false, b, (short)0, c, false, false);
}

// A-fragment (bf16 A, 16x32): lane holds row m=l&15; element e -> k = e + 16*hi
__device__ __forceinline__ v16bf load_a_frag_bf(const __bf16* row, int k0, int hi) {
  const __bf16* p = row + k0 + hi * 16;
  v8bf lo  = *(const v8bf*)(p);
  v8bf hi8 = *(const v8bf*)(p + 8);
  v16bf a;
#pragma unroll
  for (int e = 0; e < 8; ++e) { a[e] = lo[e]; a[8 + e] = hi8[e]; }
  return a;
}

// B-fragment (bf16 B, 32x16): lane holds col n=l&15; element e -> k = e + 16*hi
__device__ __forceinline__ v16bf load_b_frag_bf(const __bf16* wrow, int k0, int hi) {
  return *(const v16bf*)(wrow + k0 + hi * 16);
}

__device__ __forceinline__ float sigmoidf_(float x) {
  return 1.0f / (1.0f + __expf(-x));
}

// ---- Kernel 0: zero outputs + cum ------------------------------------------

__global__ void zero_kernel(float* __restrict__ out, int n_out, float* __restrict__ cum) {
  int i = blockIdx.x * blockDim.x + threadIdx.x;
  int stride = gridDim.x * blockDim.x;
  for (int k = i; k < n_out; k += stride) out[k] = 0.0f;
  for (int k = i; k < B_SZ; k += stride) cum[k] = 0.0f;
}

// ---- Kernel 0b: f32 -> bf16 conversion (weights, initial hidden) -----------

__global__ void cvt_bf16_kernel(const float* __restrict__ in, __bf16* __restrict__ out, int n) {
  int i = (blockIdx.x * blockDim.x + threadIdx.x) * 4;
  int stride = gridDim.x * blockDim.x * 4;
  for (int k = i; k < n; k += stride) {
    float4 v = *(const float4*)(in + k);
    v4bf b;
    b[0] = (__bf16)v.x; b[1] = (__bf16)v.y; b[2] = (__bf16)v.z; b[3] = (__bf16)v.w;
    *(v4bf*)(out + k) = b;
  }
}

// ---- Kernel 0c: E[l][o] = b_ih[o] + emb_l . w_ih_row_o  (10 x 3H, once) ----

__global__ void emb_gemm_kernel(const float* __restrict__ demb, const float* __restrict__ w_ih,
                                const float* __restrict__ b_ih, float* __restrict__ E) {
  int idx = blockIdx.x * blockDim.x + threadIdx.x;
  if (idx >= L_SZ * 3 * H_SZ) return;
  int l = idx / (3 * H_SZ);
  int o = idx % (3 * H_SZ);
  const float* e = demb + (size_t)l * H_SZ;
  const float* w = w_ih + (size_t)o * H_SZ;
  float acc = 0.0f;
  for (int k = 0; k < H_SZ; k += 4) {
    float4 ev = *(const float4*)(e + k);
    float4 wv = *(const float4*)(w + k);
    acc += ev.x * wv.x + ev.y * wv.y + ev.z * wv.z + ev.w * wv.w;
  }
  E[idx] = acc + b_ih[o];
}

// ---- Kernel 1: fused halting MLP + ACT bookkeeping -------------------------

__global__ __launch_bounds__(128) void halt_act_kernel(
    const __bf16* __restrict__ hbf_g, const __bf16* __restrict__ w1_bf,
    const float* __restrict__ b1, const float* __restrict__ w2,
    const float* __restrict__ b2,
    float* __restrict__ cum, float* __restrict__ p_step,
    float* __restrict__ acc_halt, float* __restrict__ num_upd) {
  __shared__ __bf16 h_bf[16 * H_SZ];
  __shared__ float  psum[16];

  const int tid  = threadIdx.x;
  const int lane = tid & 31;
  const int wave = tid >> 5;
  const int m0   = blockIdx.x * 16;

  stage_h_bf(hbf_g + (size_t)m0 * H_SZ, h_bf, tid);
  if (tid < 16) psum[tid] = 0.0f;
  __syncthreads();

  const int n  = lane & 15;
  const int hi = lane >> 4;
  const __bf16* arow = h_bf + (lane & 15) * H_SZ;

  float pacc[8];
#pragma unroll
  for (int v = 0; v < 8; ++v) pacc[v] = 0.0f;

  for (int j = wave * 16; j < H_SZ; j += 64) {
    v8f c = {0.f, 0.f, 0.f, 0.f, 0.f, 0.f, 0.f, 0.f};
    const __bf16* wrow = w1_bf + (size_t)(j + n) * H_SZ;
    for (int k0 = 0; k0 < H_SZ; k0 += 32) {
      v16bf a = load_a_frag_bf(arow, k0, hi);
      v16bf b = load_b_frag_bf(wrow, k0, hi);
      c = wmma_bf16(a, b, c);
    }
    float bias = b1[j + n];
    float wv   = w2[j + n];
#pragma unroll
    for (int v = 0; v < 8; ++v) {
      float t = c[v] + bias;
      t = t > 0.0f ? t : 0.0f;
      pacc[v] += t * wv;
    }
  }

#pragma unroll
  for (int off = 8; off >= 1; off >>= 1) {
#pragma unroll
    for (int v = 0; v < 8; ++v) pacc[v] += __shfl_xor(pacc[v], off, 32);
  }
  if ((lane & 15) == 0) {
#pragma unroll
    for (int v = 0; v < 8; ++v) atomicAdd(&psum[v + 8 * hi], pacc[v]);
  }
  __syncthreads();

  if (tid < 16) {
    int b = m0 + tid;
    float pv     = sigmoidf_(psum[tid] + b2[0]);
    float c      = cum[b];
    float alive  = (c < 1.0f) ? 1.0f : 0.0f;
    float nh     = (pv * alive + c > THRESH) ? 1.0f : 0.0f;
    float rem    = 1.0f - c + ACT_EPS;
    float alive2 = (1.0f - nh) * alive;
    float ps     = pv * alive2 + rem * nh;
    cum[b]       = c + ps;
    p_step[b]    = ps;
    acc_halt[b] += ps * alive2;
    num_upd[b]  += alive2;
  }
}

// ---- Kernel 2: fused GRU cell + merged accumulation ------------------------
// gi = h.W_ih^T + E[l]  (E folds emb.W_ih^T + b_ih), gh = h.W_hh^T + b_hh.
// Single A matrix; in-place f32 h update + persistent bf16 shadow copy.

__global__ __launch_bounds__(128) void gru_kernel(
    float* __restrict__ h, __bf16* __restrict__ hbf_g,
    const __bf16* __restrict__ wih_bf, const __bf16* __restrict__ whh_bf,
    const float* __restrict__ Erow,  // E + l*3H
    const float* __restrict__ b_hh,
    const float* __restrict__ p_step, float* __restrict__ merged) {
  __shared__ __bf16 h_bf[16 * H_SZ];

  const int tid  = threadIdx.x;
  const int lane = tid & 31;
  const int wave = tid >> 5;
  const int m0   = blockIdx.x * 16;

  stage_h_bf(hbf_g + (size_t)m0 * H_SZ, h_bf, tid);
  __syncthreads();

  const int n  = lane & 15;
  const int hi = lane >> 4;
  const __bf16* hrow = h_bf + (lane & 15) * H_SZ;

  for (int j = wave * 16; j < H_SZ; j += 64) {
    v8f cir = {0,0,0,0,0,0,0,0}, ciz = {0,0,0,0,0,0,0,0}, cin = {0,0,0,0,0,0,0,0};
    v8f chr_ = {0,0,0,0,0,0,0,0}, chz = {0,0,0,0,0,0,0,0}, chn = {0,0,0,0,0,0,0,0};

    const __bf16* wi_r = wih_bf + (size_t)(j + n) * H_SZ;
    const __bf16* wi_z = wi_r + (size_t)H_SZ * H_SZ;
    const __bf16* wi_n = wi_z + (size_t)H_SZ * H_SZ;
    const __bf16* wh_r = whh_bf + (size_t)(j + n) * H_SZ;
    const __bf16* wh_z = wh_r + (size_t)H_SZ * H_SZ;
    const __bf16* wh_n = wh_z + (size_t)H_SZ * H_SZ;

    for (int k0 = 0; k0 < H_SZ; k0 += 32) {
      v16bf ah = load_a_frag_bf(hrow, k0, hi);
      v16bf b;
      b = load_b_frag_bf(wi_r, k0, hi); cir  = wmma_bf16(ah, b, cir);
      b = load_b_frag_bf(wi_z, k0, hi); ciz  = wmma_bf16(ah, b, ciz);
      b = load_b_frag_bf(wi_n, k0, hi); cin  = wmma_bf16(ah, b, cin);
      b = load_b_frag_bf(wh_r, k0, hi); chr_ = wmma_bf16(ah, b, chr_);
      b = load_b_frag_bf(wh_z, k0, hi); chz  = wmma_bf16(ah, b, chz);
      b = load_b_frag_bf(wh_n, k0, hi); chn  = wmma_bf16(ah, b, chn);
    }

    const float bir = Erow[j + n], biz = Erow[H_SZ + j + n], bin = Erow[2 * H_SZ + j + n];
    const float bhr = b_hh[j + n], bhz = b_hh[H_SZ + j + n], bhn = b_hh[2 * H_SZ + j + n];

#pragma unroll
    for (int v = 0; v < 8; ++v) {
      int mloc = v + 8 * hi;
      size_t gidx = (size_t)(m0 + mloc) * H_SZ + j + n;
      float r  = sigmoidf_(cir[v] + bir + chr_[v] + bhr);
      float z  = sigmoidf_(ciz[v] + biz + chz[v] + bhz);
      float ng = tanhf(cin[v] + bin + r * (chn[v] + bhn));
      float hold = h[gidx];                  // own element, f32 exact
      float hnew = (1.0f - z) * ng + z * hold;
      h[gidx]     = hnew;                    // in-place f32 state
      hbf_g[gidx] = (__bf16)hnew;            // bf16 shadow for next step's staging
      float ps = p_step[m0 + mloc];
      merged[gidx] += ps * hold;             // MEAN_FIELD merge with pre-update h
    }
  }
}

// ---- Host launch -----------------------------------------------------------

extern "C" void kernel_launch(void* const* d_in, const int* in_sizes, int n_in,
                              void* d_out, int out_size, void* d_ws, size_t ws_size,
                              hipStream_t stream) {
  (void)in_sizes; (void)n_in; (void)out_size; (void)ws_size;

  const float* hidden0   = (const float*)d_in[1];
  const float* halt_w1   = (const float*)d_in[2];
  const float* halt_b1   = (const float*)d_in[3];
  const float* halt_w2   = (const float*)d_in[4];
  const float* halt_b2   = (const float*)d_in[5];
  const float* w_ih      = (const float*)d_in[6];
  const float* w_hh      = (const float*)d_in[7];
  const float* b_ih      = (const float*)d_in[8];
  const float* b_hh      = (const float*)d_in[9];
  const float* depth_emb = (const float*)d_in[10];

  float* merged   = (float*)d_out;                        // B*H
  float* acc_halt = merged + (size_t)B_SZ * H_SZ;         // B
  float* num_upd  = acc_halt + B_SZ;                      // B

  float* h     = (float*)d_ws;                            // B*H f32 state
  float* cum   = h + (size_t)B_SZ * H_SZ;                 // B
  float* pstep = cum + B_SZ;                              // B
  float* E     = pstep + B_SZ;                            // L*3H (emb.W_ih^T + b_ih)
  __bf16* w1_bf  = (__bf16*)(E + L_SZ * 3 * H_SZ);        // H*H
  __bf16* wih_bf = w1_bf + (size_t)H_SZ * H_SZ;           // 3H*H
  __bf16* whh_bf = wih_bf + (size_t)3 * H_SZ * H_SZ;      // 3H*H
  __bf16* hbf_g  = whh_bf + (size_t)3 * H_SZ * H_SZ;      // B*H bf16 shadow of h

  (void)hipMemcpyAsync(h, hidden0, (size_t)B_SZ * H_SZ * sizeof(float),
                       hipMemcpyDeviceToDevice, stream);
  zero_kernel<<<512, 256, 0, stream>>>(merged, B_SZ * H_SZ + 2 * B_SZ, cum);

  // one-time conversions / precompute
  cvt_bf16_kernel<<<256, 256, 0, stream>>>(halt_w1, w1_bf, H_SZ * H_SZ);
  cvt_bf16_kernel<<<512, 256, 0, stream>>>(w_ih, wih_bf, 3 * H_SZ * H_SZ);
  cvt_bf16_kernel<<<512, 256, 0, stream>>>(w_hh, whh_bf, 3 * H_SZ * H_SZ);
  cvt_bf16_kernel<<<2048, 256, 0, stream>>>(hidden0, hbf_g, B_SZ * H_SZ);
  emb_gemm_kernel<<<(L_SZ * 3 * H_SZ + 255) / 256, 256, 0, stream>>>(depth_emb, w_ih, b_ih, E);

  for (int l = 0; l < L_SZ; ++l) {
    halt_act_kernel<<<B_SZ / 16, 128, 0, stream>>>(hbf_g, w1_bf, halt_b1, halt_w2, halt_b2,
                                                   cum, pstep, acc_halt, num_upd);
    gru_kernel<<<B_SZ / 16, 128, 0, stream>>>(h, hbf_g, wih_bf, whh_bf,
                                              E + (size_t)l * 3 * H_SZ, b_hh,
                                              pstep, merged);
  }
}
